// RegressorLinewise_38749194944900
// MI455X (gfx1250) — matrix operand order repaired and checked by reading, not compiled, not run on hardware
//
#include <hip/hip_runtime.h>
#include <hip/hip_bf16.h>

#define LINES 448
#define WID   128
#define WT    32          // w-columns handled per workgroup
#define NL    6
#define BSTR  1032        // padded LDS row stride in bf16 elems (x2B: 16B-aligned rows)
#define MAXC  1024

typedef __attribute__((ext_vector_type(4)))  float  v4f;
typedef __attribute__((ext_vector_type(8)))  float  v8f;
typedef __attribute__((ext_vector_type(8)))  __bf16 v8bf;
typedef __attribute__((ext_vector_type(16))) __bf16 v16bf;

struct KArgs {
  const float* x;
  const float* Wt[NL];
  const float* bi[NL];
  const float* ga[NL];
  const float* be[NL];
  const float* mu[NL];
  const float* va[NL];
  const float* ow;
  const float* ob;
  float* out;
};

struct BfPair { __bf16 h, l; };

// f = hi + lo with hi = bf16(f); residual is exact in fp32.
__device__ __forceinline__ BfPair split_bf16(float f) {
  BfPair p;
  p.h = (__bf16)f;
  p.l = (__bf16)(f - (float)p.h);
  return p;
}

// 16 packed bf16 from LDS as two 16B loads (rows are 16B- but not 32B-aligned).
__device__ __forceinline__ v16bf ld_bf16x16(const __bf16* p) {
  v8bf a = *(const v8bf*)p;
  v8bf b = *(const v8bf*)(p + 8);
  return __builtin_shufflevector(a, b, 0, 1, 2, 3, 4, 5, 6, 7,
                                 8, 9, 10, 11, 12, 13, 14, 15);
}

__device__ __forceinline__ v8f wmma_bf16(v16bf a, v16bf b, v8f c) {
  return __builtin_amdgcn_wmma_f32_16x16x32_bf16(
      false, a, false, b, (short)0, c, false, false);
}

__global__ __launch_bounds__(256, 1) void regressor_linewise(KArgs a) {
  constexpr int chs[NL + 1] = {128, 64, 64, 128, 1024, 1024, 128};

  extern __shared__ char smem_raw[];
  __bf16* planeAh = (__bf16*)smem_raw;         // activations hi, buffer A  [WT][BSTR]
  __bf16* planeAl = planeAh + WT * BSTR;       // activations lo, buffer A
  __bf16* planeBh = planeAl + WT * BSTR;       // hi, buffer B
  __bf16* planeBl = planeBh + WT * BSTR;       // lo, buffer B
  float*  scl     = (float*)(planeBl + WT * BSTR);  // [MAXC] fused BN scale
  float*  shf     = scl + MAXC;                     // [MAXC] fused BN shift

  const int line = blockIdx.y;
  const int w0   = blockIdx.x * WT;
  const int tid  = threadIdx.x;
  const int wave = tid >> 5;
  const int lane = tid & 31;
  const int m    = lane & 15;   // M index (A) / N index (B,D)
  const int g2   = lane >> 4;   // lane-group (K offset 8*g2 / 16*g2; D row offset 8*g2)

  // Stage x tile transposed + split: plane[w][ci] = split(x[0, ci, line, w0+w])
  for (int i = tid; i < chs[0] * WT; i += 256) {
    int ci = i >> 5;
    int w  = i & (WT - 1);
    float f = a.x[((size_t)ci * LINES + line) * WID + w0 + w];
    BfPair p = split_bf16(f);
    planeAh[w * BSTR + ci] = p.h;
    planeAl[w * BSTR + ci] = p.l;
  }

  __bf16* hinh = planeAh;
  __bf16* hinl = planeAl;
  __bf16* houth = planeBh;
  __bf16* houtl = planeBl;

  for (int L = 0; L < NL; ++L) {
    const int CI = chs[L], CO = chs[L + 1];

    // Fused BN(eval)+bias:  y = d*scl[c] + shf[c]
    for (int c = tid; c < CO; c += 256) {
      size_t o = (size_t)line * CO + c;
      float s = a.ga[L][o] * rsqrtf(a.va[L][o] + 1e-5f);
      scl[c] = s;
      shf[c] = (a.bi[L][o] - a.mu[L][o]) * s + a.be[L][o];
    }
    __syncthreads();

    const float* Wl  = a.Wt[L] + (size_t)line * CO * CI;
    const int    nrt = CO >> 4;                 // 16-row output tiles

    for (int rt = wave; rt < nrt; rt += 8) {
      const float* wrow = Wl + (size_t)(rt * 16 + m) * CI;  // this lane's weight row
      v8f acc0 = {};
      v8f acc1 = {};
      for (int k0 = 0; k0 < CI; k0 += 32) {
        // ---- A fragment: 16 fp32 per lane, split to bf16 hi/lo in registers.
        // 16-bit A layout: lane (m,g2): K = 16*g2 + 0..15, contiguous (matches B)
        const float* wp = wrow + k0 + 16 * g2;
        __builtin_prefetch(wrow + k0 + 512, 0, 0);   // stream weights from HBM
        v4f fa = *(const v4f*)(wp);
        v4f fb = *(const v4f*)(wp + 4);
        v4f fc = *(const v4f*)(wp + 8);
        v4f fd = *(const v4f*)(wp + 12);
        v16bf ah, al;
        #pragma unroll
        for (int e = 0; e < 4; ++e) {
          BfPair pa = split_bf16(fa[e]);
          BfPair pb = split_bf16(fb[e]);
          BfPair pc = split_bf16(fc[e]);
          BfPair pd = split_bf16(fd[e]);
          ah[e]      = pa.h; al[e]      = pa.l;
          ah[4 + e]  = pb.h; al[4 + e]  = pb.l;
          ah[8 + e]  = pc.h; al[8 + e]  = pc.l;
          ah[12 + e] = pd.h; al[12 + e] = pd.l;
        }
        // ---- B fragments from LDS bf16 planes.
        // 16-bit B layout: lane (n=m|16+m, g2): K = 16*g2 + 0..15, contiguous in [n][k]
        const int boff = k0 + 16 * g2;
        v16bf b0h = ld_bf16x16(hinh + m * BSTR + boff);
        v16bf b0l = ld_bf16x16(hinl + m * BSTR + boff);
        v16bf b1h = ld_bf16x16(hinh + (16 + m) * BSTR + boff);
        v16bf b1l = ld_bf16x16(hinl + (16 + m) * BSTR + boff);
        // ---- bf16x3 split product: Ah*Bh + Ah*Bl + Al*Bh (fp32 accumulate)
        acc0 = wmma_bf16(ah, b0h, acc0);
        acc1 = wmma_bf16(ah, b1h, acc1);
        acc0 = wmma_bf16(ah, b0l, acc0);
        acc1 = wmma_bf16(ah, b1l, acc1);
        acc0 = wmma_bf16(al, b0h, acc0);
        acc1 = wmma_bf16(al, b1h, acc1);
      }
      // ---- Epilogue: BN + LeakyReLU, re-split to bf16 hi/lo planes, transposed store.
      const int cb = rt * 16 + 8 * g2;       // 8 consecutive channels per lane
      v8bf yh0, yl0, yh1, yl1;
      #pragma unroll
      for (int v = 0; v < 8; ++v) {
        float s  = scl[cb + v];
        float t  = shf[cb + v];
        float y0 = acc0[v] * s + t;
        float y1 = acc1[v] * s + t;
        y0 = (y0 >= 0.f) ? y0 : 0.01f * y0;
        y1 = (y1 >= 0.f) ? y1 : 0.01f * y1;
        BfPair p0 = split_bf16(y0);
        BfPair p1 = split_bf16(y1);
        yh0[v] = p0.h; yl0[v] = p0.l;
        yh1[v] = p1.h; yl1[v] = p1.l;
      }
      *(v8bf*)(houth + m * BSTR + cb)        = yh0;   // 16B stores, 16B-aligned
      *(v8bf*)(houtl + m * BSTR + cb)        = yl0;
      *(v8bf*)(houth + (16 + m) * BSTR + cb) = yh1;
      *(v8bf*)(houtl + (16 + m) * BSTR + cb) = yl1;
    }
    __syncthreads();
    __bf16* t;
    t = hinh; hinh = houth; houth = t;
    t = hinl; hinl = houtl; houtl = t;
  }

  // Output head: per w, dot(out_w[line], h[:,w]) + out_b[line]
  if (tid < WT) {
    const __bf16* hh = hinh + tid * BSTR;
    const __bf16* hl = hinl + tid * BSTR;
    const float* ow = a.ow + (size_t)line * chs[NL];
    float acc = 0.f;
    #pragma unroll 8
    for (int ci = 0; ci < chs[NL]; ++ci)
      acc += ow[ci] * ((float)hh[ci] + (float)hl[ci]);
    a.out[(size_t)line * WID + w0 + tid] = acc + a.ob[line];
  }
}

extern "C" void kernel_launch(void* const* d_in, const int* in_sizes, int n_in,
                              void* d_out, int out_size, void* d_ws, size_t ws_size,
                              hipStream_t stream) {
  (void)in_sizes; (void)n_in; (void)d_ws; (void)ws_size; (void)out_size;

  KArgs a;
  a.x = (const float*)d_in[0];
  for (int l = 0; l < NL; ++l) {
    a.Wt[l] = (const float*)d_in[1 + 6 * l + 0];
    a.bi[l] = (const float*)d_in[1 + 6 * l + 1];
    a.ga[l] = (const float*)d_in[1 + 6 * l + 2];
    a.be[l] = (const float*)d_in[1 + 6 * l + 3];
    a.mu[l] = (const float*)d_in[1 + 6 * l + 4];
    a.va[l] = (const float*)d_in[1 + 6 * l + 5];
  }
  a.ow  = (const float*)d_in[1 + 6 * NL];
  a.ob  = (const float*)d_in[2 + 6 * NL];
  a.out = (float*)d_out;

  const int smem_bytes = 4 * WT * BSTR * (int)sizeof(__bf16)
                       + 2 * MAXC * (int)sizeof(float);   // ~266 KB < 320 KB/WGP
  (void)hipFuncSetAttribute((const void*)regressor_linewise,
                            hipFuncAttributeMaxDynamicSharedMemorySize, smem_bytes);

  dim3 grid(WID / WT, LINES);
  regressor_linewise<<<grid, 256, smem_bytes, stream>>>(a);
}
